// PNN_85701777424456
// MI455X (gfx1250) — matrix operand order repaired and checked by reading, not compiled, not run on hardware
//
#include <hip/hip_runtime.h>
#include <hip/hip_bf16.h>

#define B_    4096
#define N_    50
#define M_    64
#define D_    64
#define NM_   3200     // N_*M_
#define MM_   4096     // M_*M_
#define BN_EPS 1e-3f

typedef __attribute__((ext_vector_type(16))) __bf16 v16bf;
typedef __attribute__((ext_vector_type(8)))  float  v8f;
typedef __attribute__((ext_vector_type(4)))  int    v4i_;

union FragU { v16bf v; uint4 q[2]; };

#ifndef __has_builtin
#define __has_builtin(x) 0
#endif

#if __has_builtin(__builtin_amdgcn_global_load_async_to_lds_b128) && \
    __has_builtin(__builtin_amdgcn_s_wait_asynccnt)
#define USE_ASYNC_LDS 1
#else
#define USE_ASYNC_LDS 0
#endif

#if USE_ASYNC_LDS
// per-lane 16B global->LDS async copy (tracked by ASYNCcnt)
__device__ __forceinline__ void async_cp16(const void* g, void* l) {
    typedef __attribute__((address_space(1))) v4i_ gv4_t;   // global int4
    typedef __attribute__((address_space(3))) v4i_ lv4_t;   // LDS int4
    __builtin_amdgcn_global_load_async_to_lds_b128(
        (gv4_t*)(uintptr_t)g,
        (lv4_t*)(unsigned)(uintptr_t)l,   // flat LDS addr low 32b == LDS offset
        0, 0);
}
#endif

// ---------------------------------------------------------------------------
// f32 -> bf16 copy (weights already in [Ncols, K] layout)
__global__ void k_cvt_bf16(const float* __restrict__ src,
                           __hip_bfloat16* __restrict__ dst, int n) {
    int i = blockIdx.x * blockDim.x + threadIdx.x;
    if (i < n) dst[i] = __float2bfloat16(src[i]);
}

// f32 [rows,cols] -> bf16 [cols,rows]  (MLP weights need the transpose)
__global__ void k_cvt_bf16_t(const float* __restrict__ src,
                             __hip_bfloat16* __restrict__ dst,
                             int rows, int cols) {
    int i = blockIdx.x * blockDim.x + threadIdx.x;
    if (i < rows * cols) {
        int r = i / cols, c = i - r * cols;
        dst[(size_t)c * rows + r] = __float2bfloat16(src[i]);
    }
}

// ---------------------------------------------------------------------------
// Gather + scale: feats[b,n,m] = emb[idx[b,n],m]*fv[b,n]  (bf16 out)
// plus s[b,m] = sum_n feats, feats2[b,n] = sum_m feats^2
__global__ void k_gather(const float* __restrict__ fv,
                         const int*   __restrict__ idx,
                         const float* __restrict__ emb,
                         __hip_bfloat16* __restrict__ feats,
                         float* __restrict__ s_out,
                         float* __restrict__ feats2) {
    int b = blockIdx.x;          // one block per batch row
    int m = threadIdx.x;         // 64 threads = 2 waves
    __shared__ float f2[N_];
    if (m < N_) f2[m] = 0.f;
    __syncthreads();
    float s_acc = 0.f;
    for (int n = 0; n < N_; ++n) {
        int   id = idx[b * N_ + n];
        float v  = fv[b * N_ + n];
        float f  = emb[(size_t)id * M_ + m] * v;
        feats[(size_t)b * NM_ + n * M_ + m] = __float2bfloat16(f);
        s_acc += f;
        atomicAdd(&f2[n], f * f);            // ds_add_f32
    }
    s_out[(size_t)b * M_ + m] = s_acc;
    __syncthreads();
    if (m < N_) feats2[(size_t)b * N_ + m] = f2[m];
}

// ---------------------------------------------------------------------------
// Outer product P[b, m*64+n] = s[b,m]*s[b,n] (bf16)
__global__ void k_outer(const float* __restrict__ s,
                        __hip_bfloat16* __restrict__ P) {
    int b = blockIdx.x;
    __shared__ float sl[M_];
    if (threadIdx.x < M_) sl[threadIdx.x] = s[(size_t)b * M_ + threadIdx.x];
    __syncthreads();
    for (int i = threadIdx.x; i < MM_; i += blockDim.x)
        P[(size_t)b * MM_ + i] = __float2bfloat16(sl[i >> 6] * sl[i & 63]);
}

// ---------------------------------------------------------------------------
// l_p_in[b,d] = sum_n feats2[b,n] * theta[d,n]^2  -> xcat col 64+d (bf16)
__global__ void k_lp_in(const float* __restrict__ feats2,
                        const float* __restrict__ theta,
                        __hip_bfloat16* __restrict__ xcat) {
    int t = blockIdx.x * blockDim.x + threadIdx.x;
    if (t >= B_ * D_) return;
    int b = t >> 6, d = t & 63;
    float acc = 0.f;
    for (int n = 0; n < N_; ++n) {
        float th = theta[d * N_ + n];
        acc += feats2[(size_t)b * N_ + n] * th * th;
    }
    xcat[(size_t)b * 192 + 64 + d] = __float2bfloat16(acc);
}

// ---------------------------------------------------------------------------
// bf16 WMMA GEMM, register-blocked 16x64 per wave, 64x64 per 4-wave block.
// B panel (64 rows x 32 K, 4KB) is staged in LDS, double-buffered via
// GLOBAL_LOAD_ASYNC_TO_LDS_B128 + s_wait_asynccnt (shared by all 4 waves).
// Out[r, oco+j] = sum_k A[r,k]*BT[j,k]; optional bias+BN+ReLU epilogue.
// Requires Mrows%64==0, Ncols%64==0, K%32==0; full EXEC (WMMA req).
__global__ void k_gemm_bf16(const __hip_bfloat16* __restrict__ A,
                            const __hip_bfloat16* __restrict__ BT,
                            __hip_bfloat16* __restrict__ Out,
                            int Mrows, int Ncols, int K,
                            int outStride, int outColOff,
                            const float* __restrict__ bias,
                            const float* __restrict__ gamma,
                            const float* __restrict__ beta,
                            const float* __restrict__ mu,
                            const float* __restrict__ var) {
    const int nBlk = Ncols >> 6;               // 64-col blocks
    const int bm   = blockIdx.x / nBlk;
    const int bn   = blockIdx.x - bm * nBlk;
    const int nc0  = bn * 64;
    const int wid  = threadIdx.x >> 5;         // wave 0..3 -> 16-row strip
    const int lane = threadIdx.x & 31;
    const int r    = lane & 15;
    const int hi   = lane >> 4;                // 0: lanes 0-15, 1: lanes 16-31
    const int row0 = bm * 64 + wid * 16;

    __shared__ __align__(16) __hip_bfloat16 panel[2][64 * 32];  // 2 x 4KB

    const __hip_bfloat16* arow = A + (size_t)(row0 + r) * K;

    // B-panel staging: 256 16B chunks, 128 threads -> 2 chunks each
    auto issue = [&](int buf, int k0) {
        #pragma unroll
        for (int cc = 0; cc < 2; ++cc) {
            int c   = (int)threadIdx.x + cc * 128;
            int row = c >> 2;                  // panel row (0..63)
            int ko  = (c & 3) * 8;             // K offset within panel (elems)
            const __hip_bfloat16* g = BT + (size_t)(nc0 + row) * K + k0 + ko;
            __hip_bfloat16* l = &panel[buf][row * 32 + ko];
#if USE_ASYNC_LDS
            async_cp16(g, l);
#else
            *(uint4*)l = *(const uint4*)g;
#endif
        }
    };

    v8f acc[4];
    #pragma unroll
    for (int nb = 0; nb < 4; ++nb)
        #pragma unroll
        for (int v = 0; v < 8; ++v) acc[nb][v] = 0.f;

    int cur = 0;
    issue(0, 0);
    for (int k0 = 0; k0 < K; k0 += 32) {
        const bool more = (k0 + 32) < K;
        if (more) {
            issue(cur ^ 1, k0 + 32);           // prefetch next panel into LDS
            __builtin_prefetch(arow + k0 + 32, 0, 3);
#if USE_ASYNC_LDS
            __builtin_amdgcn_s_wait_asynccnt(2);   // retire current panel only
#endif
        } else {
#if USE_ASYNC_LDS
            __builtin_amdgcn_s_wait_asynccnt(0);
#endif
        }
        __syncthreads();                       // panel[cur] visible to block

        FragU a;
        // A 16x32 bf16: lane<16 -> K {0..7,16..23}; lane>=16 -> K {8..15,24..31}
        a.q[0] = *(const uint4*)(arow + k0 + hi * 8);
        a.q[1] = *(const uint4*)(arow + k0 + 16 + hi * 8);

        #pragma unroll
        for (int nb = 0; nb < 4; ++nb) {
            FragU b;
            // B 32x16 bf16: lane holds col=lane&15; lane<16 -> K {0..7,16..23}; lane>=16 -> K {8..15,24..31}
            const __hip_bfloat16* lp = &panel[cur][(nb * 16 + r) * 32 + hi * 8];
            b.q[0] = *(const uint4*)lp;
            b.q[1] = *(const uint4*)(lp + 16);
            acc[nb] = __builtin_amdgcn_wmma_f32_16x16x32_bf16(
                          false, a.v, false, b.v, (short)0, acc[nb], false, false);
        }
        __syncthreads();                       // all reads of panel[cur] done
        cur ^= 1;
    }

    // epilogue: optional bias + BN(inference) + ReLU, bf16 store
    const bool ep = (bias != nullptr);
    const int  rr = row0 + hi * 8;
    #pragma unroll
    for (int nb = 0; nb < 4; ++nb) {
        int j = nc0 + nb * 16 + (lane & 15);   // output column within Ncols
        float sc = 1.f, sh = 0.f;
        if (ep) {
            float inv = rsqrtf(var[j] + BN_EPS);
            sc = gamma[j] * inv;
            sh = beta[j] + sc * (bias[j] - mu[j]);
        }
        #pragma unroll
        for (int v = 0; v < 8; ++v) {
            float x = acc[nb][v];
            if (ep) { x = sc * x + sh; x = x > 0.f ? x : 0.f; }
            Out[(size_t)(rr + v) * outStride + outColOff + j] = __float2bfloat16(x);
        }
    }
}

// ---------------------------------------------------------------------------
// out[b] = sigmoid(h2[b,:] . Wout + bout)
__global__ void k_final(const __hip_bfloat16* __restrict__ h2,
                        const float* __restrict__ Wout,
                        const float* __restrict__ bout,
                        float* __restrict__ out) {
    int b = blockIdx.x;
    int j = threadIdx.x;                       // 128 threads
    __shared__ float red[128];
    red[j] = __bfloat162float(h2[(size_t)b * 128 + j]) * Wout[j];
    __syncthreads();
    for (int s = 64; s > 0; s >>= 1) {
        if (j < s) red[j] += red[j + s];
        __syncthreads();
    }
    if (j == 0) out[b] = 1.f / (1.f + expf(-(red[0] + bout[0])));
}

// ---------------------------------------------------------------------------
extern "C" void kernel_launch(void* const* d_in, const int* in_sizes, int n_in,
                              void* d_out, int out_size, void* d_ws, size_t ws_size,
                              hipStream_t stream) {
    const float* fv    = (const float*)d_in[0];
    const int*   idx   = (const int*)  d_in[1];
    const float* emb   = (const float*)d_in[2];
    const float* ls    = (const float*)d_in[3];   // [64,3200] == BT layout for l_z
    const float* theta = (const float*)d_in[4];
    const float* qw    = (const float*)d_in[5];   // [64,4096] == BT layout for l_p_out
    const float* W0 = (const float*)d_in[6],  *b0 = (const float*)d_in[7];
    const float* g0 = (const float*)d_in[8],  *be0 = (const float*)d_in[9];
    const float* m0 = (const float*)d_in[10], *v0 = (const float*)d_in[11];
    const float* W1 = (const float*)d_in[12], *b1 = (const float*)d_in[13];
    const float* g1 = (const float*)d_in[14], *be1 = (const float*)d_in[15];
    const float* m1 = (const float*)d_in[16], *v1 = (const float*)d_in[17];
    const float* W2 = (const float*)d_in[18], *b2 = (const float*)d_in[19];
    const float* g2 = (const float*)d_in[20], *be2 = (const float*)d_in[21];
    const float* m2 = (const float*)d_in[22], *v2 = (const float*)d_in[23];
    const float* Wout = (const float*)d_in[24], *bout = (const float*)d_in[25];
    float* out = (float*)d_out;

    // ---- workspace carve-up (bf16 intermediates; P aliases feats) ----
    char* p = (char*)d_ws;
    auto alloc = [&](size_t bytes) -> char* {
        char* r = p; p += (bytes + 255) & ~(size_t)255; return r;
    };
    __hip_bfloat16* feats = (__hip_bfloat16*)alloc((size_t)B_ * MM_ * 2); // feats then P
    __hip_bfloat16* P     = feats;
    float* s_f32  = (float*)alloc((size_t)B_ * M_ * 4);
    float* feats2 = (float*)alloc((size_t)B_ * N_ * 4);
    __hip_bfloat16* xcat = (__hip_bfloat16*)alloc((size_t)B_ * 192 * 2);
    __hip_bfloat16* h0   = (__hip_bfloat16*)alloc((size_t)B_ * 512 * 2);
    __hip_bfloat16* h1   = (__hip_bfloat16*)alloc((size_t)B_ * 256 * 2);
    __hip_bfloat16* h2   = (__hip_bfloat16*)alloc((size_t)B_ * 128 * 2);
    __hip_bfloat16* LSt  = (__hip_bfloat16*)alloc((size_t)D_ * NM_ * 2);
    __hip_bfloat16* QT   = (__hip_bfloat16*)alloc((size_t)D_ * MM_ * 2);
    __hip_bfloat16* W0T  = (__hip_bfloat16*)alloc((size_t)512 * 192 * 2);
    __hip_bfloat16* W1T  = (__hip_bfloat16*)alloc((size_t)256 * 512 * 2);
    __hip_bfloat16* W2T  = (__hip_bfloat16*)alloc((size_t)128 * 256 * 2);

    auto gemm = [&](const __hip_bfloat16* A, const __hip_bfloat16* BT,
                    __hip_bfloat16* O, int Mr, int Nc, int K, int os, int oco,
                    const float* bi, const float* ga, const float* be,
                    const float* mu, const float* va) {
        int grid = (Mr / 64) * (Nc / 64);      // 64x64 tile per 4-wave block
        k_gemm_bf16<<<grid, 128, 0, stream>>>(A, BT, O, Mr, Nc, K, os, oco,
                                              bi, ga, be, mu, va);
    };

    // weight conversions (bf16, BT layout)
    k_cvt_bf16 <<<(D_ * NM_ + 255) / 256, 256, 0, stream>>>(ls, LSt, D_ * NM_);
    k_cvt_bf16 <<<(D_ * MM_ + 255) / 256, 256, 0, stream>>>(qw, QT,  D_ * MM_);
    k_cvt_bf16_t<<<(192 * 512 + 255) / 256, 256, 0, stream>>>(W0, W0T, 192, 512);
    k_cvt_bf16_t<<<(512 * 256 + 255) / 256, 256, 0, stream>>>(W1, W1T, 512, 256);
    k_cvt_bf16_t<<<(256 * 128 + 255) / 256, 256, 0, stream>>>(W2, W2T, 256, 128);

    // embedding gather + row stats
    k_gather<<<B_, M_, 0, stream>>>(fv, idx, emb, feats, s_f32, feats2);

    // l_z  -> xcat[:, 0:64]
    gemm(feats, LSt, xcat, B_, D_, NM_, 192, 0,
         nullptr, nullptr, nullptr, nullptr, nullptr);
    // l_p_in -> xcat[:, 64:128]
    k_lp_in<<<(B_ * D_ + 255) / 256, 256, 0, stream>>>(feats2, theta, xcat);
    // l_p_out: P = outer(s,s) (reuses feats buffer), then GEMM -> xcat[:, 128:192]
    k_outer<<<B_, 256, 0, stream>>>(s_f32, P);
    gemm(P, QT, xcat, B_, D_, MM_, 192, 128,
         nullptr, nullptr, nullptr, nullptr, nullptr);

    // MLP with fused bias + BN(inference) + ReLU epilogues
    gemm(xcat, W0T, h0, B_, 512, 192, 512, 0, b0, g0, be0, m0, v0);
    gemm(h0,   W1T, h1, B_, 256, 512, 256, 0, b1, g1, be1, m1, v1);
    gemm(h1,   W2T, h2, B_, 128, 256, 128, 0, b2, g2, be2, m2, v2);

    // final dot + sigmoid
    k_final<<<B_, 128, 0, stream>>>(h2, Wout, bout, out);
}
